// DenseGraphSimpleOpEdgeFlow_32091995636312
// MI455X (gfx1250) — matrix operand order repaired and checked by reading, not compiled, not run on hardware
//
#include <hip/hip_runtime.h>
#include <hip/hip_bf16.h>

typedef __attribute__((ext_vector_type(16))) _Float16 v16h;
typedef __attribute__((ext_vector_type(8)))  _Float16 v8h;
typedef __attribute__((ext_vector_type(8)))  float    v8f;
typedef __attribute__((ext_vector_type(2)))  float    v2f;

#define Bc   64
#define Nc   96
#define INF  128
#define OUTF 128
#define OPD  48
#define KPAD 64
#define ROWS 72   // padded LDS row stride in f16 elements (144B, 16B-aligned, bank-friendly)

// ---------------------------------------------------------------------------
// Kernel 1: support = inputs @ weight   ([6144,128] @ [128,128], full fp32)
// One 16x16 output tile per wave via V_WMMA_F32_16X16X4_F32, K-loop of 32 steps.
// ---------------------------------------------------------------------------
__global__ __launch_bounds__(256) void support_gemm(const float* __restrict__ in,
                                                    const float* __restrict__ w,
                                                    float* __restrict__ S) {
  const int tid  = threadIdx.x;
  const int lane = tid & 31;
  const int wv   = tid >> 5;               // 8 waves per block
  const int tile = blockIdx.x * 8 + wv;    // 384*8 = 3072 tiles
  const int mt   = tile >> 3;              // 0..383  (M tiles of 16 over 6144 rows)
  const int nt   = tile & 7;               // 0..7    (N tiles of 16 over 128 cols)
  const bool hi  = lane >= 16;
  const int l16  = lane & 15;
  const int m    = mt * 16 + l16;
  const int n    = nt * 16 + l16;

  v8f c = {};
  for (int k = 0; k < INF; k += 4) {
    // A frag (16x4 f32): lanes 0-15 hold K={k,k+1}, lanes 16-31 hold K={k+2,k+3}
    const float* ap = in + m * INF + k + (hi ? 2 : 0);
    v2f a; a.x = ap[0]; a.y = ap[1];
    // B frag (4x16 f32): same K split, column n striped across lanes
    const int kb = k + (hi ? 2 : 0);
    v2f bb; bb.x = w[kb * OUTF + n]; bb.y = w[(kb + 1) * OUTF + n];
    c = __builtin_amdgcn_wmma_f32_16x16x4_f32(false, a, false, bb, (short)0, c,
                                              false, false);
  }
#pragma unroll
  for (int r = 0; r < 8; ++r) {
    const int row = mt * 16 + r + (hi ? 8 : 0);   // D layout: lanes16-31 -> M=r+8
    S[row * OUTF + n] = c[r];
  }
}

// ---------------------------------------------------------------------------
// Kernel 2: fused edge-attention + masked aggregation + residual.
// One workgroup per (b,i); 8 waves; wave w owns features d = w*16..w*16+15.
// attn logits via v_wmma_f32_16x16x32_f16 (K=48 padded to 64 -> 2 k-steps),
// sigmoid/adj-mask applied in registers, j-reduction against support, so the
// 302 MB attn tensor is never materialized.
// ---------------------------------------------------------------------------
__global__ __launch_bounds__(256) void attn_aggregate(
    const int* __restrict__ adj, const float* __restrict__ op_emb,
    const float* __restrict__ attn_w, const float* __restrict__ attn_b,
    const float* __restrict__ self_emb, const float* __restrict__ S,
    float* __restrict__ out) {
  __shared__ __align__(16) _Float16 sA[Nc * ROWS];     // op_emb' tile [96][72]
  __shared__ __align__(16) _Float16 sW[OUTF * ROWS];   // attn_w^T     [128][72]
  __shared__ int sCode[Nc];                            // 0=masked,1=skip,2=sigmoid

  const int bi  = blockIdx.x;          // 0..6143
  const int b   = bi / Nc;
  const int i   = bi - b * Nc;
  const int tid = threadIdx.x;

  // adj' = adj + I ; classify each edge once per j
  if (tid < Nc) {
    int a = adj[(b * Nc + i) * Nc + tid] + (tid == i ? 1 : 0);
    sCode[tid] = (a == 1) ? 1 : ((a == 0) ? 0 : 2);
  }
  // Stage op_emb'[b,i] (diagonal row j==i replaced by self_op_emb), f32->f16, K padded
  for (int e = tid; e < Nc * KPAD; e += 256) {
    const int j = e >> 6, cc = e & 63;
    float v = 0.f;
    if (cc < OPD)
      v = (j == i) ? self_emb[cc] : op_emb[(((b * Nc + i) * Nc) + j) * OPD + cc];
    sA[j * ROWS + cc] = (_Float16)v;
  }
  // Stage attn_w transposed: sW[d][k]
  for (int e = tid; e < OUTF * KPAD; e += 256) {
    const int d = e >> 6, k = e & 63;
    const float v = (k < OPD) ? attn_w[k * OUTF + d] : 0.f;
    sW[d * ROWS + k] = (_Float16)v;
  }
  __syncthreads();

  const int lane = tid & 31;
  const int wv   = tid >> 5;
  const bool hi  = lane >= 16;
  const int l16  = lane & 15;
  const int d    = wv * 16 + l16;
  const float bias = attn_b[d];

  // B fragments (32x16 f16): lanes0-15 K chunks {0-7,16-23}, lanes16-31 {8-15,24-31} of column d
  v16h bf[2];
#pragma unroll
  for (int ks = 0; ks < 2; ++ks) {
    const int kb = ks * 32 + (hi ? 8 : 0);
    const v8h lo = *(const v8h*)&sW[d * ROWS + kb];
    const v8h hh = *(const v8h*)&sW[d * ROWS + kb + 16];
#pragma unroll
    for (int t = 0; t < 8; ++t) { bf[ks][t] = lo[t]; bf[ks][8 + t] = hh[t]; }
  }

  float partial = 0.f;
  const float* Sb = S + b * Nc * OUTF;
#pragma unroll
  for (int mtile = 0; mtile < 6; ++mtile) {
    const int m = mtile * 16 + l16;
    v8f c = {};
#pragma unroll
    for (int ks = 0; ks < 2; ++ks) {
      // A frag (16x32 f16): lanes0-15 K chunks {0-7,16-23}, lanes16-31 {8-15,24-31}
      const int kb = ks * 32 + (hi ? 8 : 0);
      const v8h c0 = *(const v8h*)&sA[m * ROWS + kb];
      const v8h c1 = *(const v8h*)&sA[m * ROWS + kb + 16];
      v16h a;
#pragma unroll
      for (int t = 0; t < 8; ++t) { a[t] = c0[t]; a[8 + t] = c1[t]; }
      c = __builtin_amdgcn_wmma_f32_16x16x32_f16(false, a, false, bf[ks],
                                                 (short)0, c, false, false);
    }
    // c[r] = logit(j = mtile*16 + r (+8 for hi half), d) ; mask + aggregate
#pragma unroll
    for (int r = 0; r < 8; ++r) {
      const int j = mtile * 16 + r + (hi ? 8 : 0);
      const int code = sCode[j];
      const float logit = c[r] + bias;
      const float sg = 1.f / (1.f + __expf(-logit));
      const float attn = (code == 1) ? 1.f : ((code == 0) ? 0.f : sg);
      partial += attn * Sb[j * OUTF + d];
    }
  }
  // lane l and l^16 hold the same feature d, complementary j halves
  const float total = partial + __shfl_xor(partial, 16, 32);
  if (!hi) out[(b * Nc + i) * OUTF + d] = Sb[i * OUTF + d] + total;
}

extern "C" void kernel_launch(void* const* d_in, const int* in_sizes, int n_in,
                              void* d_out, int out_size, void* d_ws, size_t ws_size,
                              hipStream_t stream) {
  const float* inputs = (const float*)d_in[0];
  const int*   adj    = (const int*)d_in[1];
  const float* op_emb = (const float*)d_in[2];
  const float* weight = (const float*)d_in[3];
  const float* attn_w = (const float*)d_in[4];
  const float* attn_b = (const float*)d_in[5];
  const float* self_e = (const float*)d_in[6];
  float* out = (float*)d_out;
  float* S   = (float*)d_ws;   // support scratch: 6144*128 f32 = 3 MB

  support_gemm<<<384, 256, 0, stream>>>(inputs, weight, S);
  attn_aggregate<<<Bc * Nc, 256, 0, stream>>>(adj, op_emb, attn_w, attn_b,
                                              self_e, S, out);
}
